// SPIDER_42734924595802
// MI455X (gfx1250) — matrix operand order repaired and checked by reading, not compiled, not run on hardware
//
#include <hip/hip_runtime.h>
#include <math.h>

// ---------------- problem constants (from reference) ----------------
#define HDIM 64
#define NHEAD 4
#define NNODES 50000
#define NEDGES 200000
#define E2 (2 * NEDGES)
#define BN_EPS 1e-5f
#define GAT_F (NHEAD * HDIM) /* 256 */

typedef __attribute__((ext_vector_type(16))) _Float16 v16h;
typedef __attribute__((ext_vector_type(8)))  _Float16 v8h;
typedef __attribute__((ext_vector_type(8)))  float    v8f;

__device__ __forceinline__ float lrelu(float x, float s) { return x >= 0.f ? x : s * x; }

__device__ __forceinline__ void atomicMaxF(float* addr, float val) {
  unsigned int* ua = (unsigned int*)addr;
  unsigned int old = *ua;
  while (__uint_as_float(old) < val) {
    unsigned int assumed = old;
    old = atomicCAS(ua, assumed, __float_as_uint(val));
    if (old == assumed) break;
  }
}

// =====================================================================
// Pre-pass 1: activations f32 -> f16, optional row gather, column slice,
// K zero-padded to Kpad (multiple of 64). Branch-free GEMM follows.
// =====================================================================
__global__ __launch_bounds__(256)
void cvt_a_f16(const float* __restrict__ A, int lda, int colOff,
               const int* __restrict__ rowIdx, int idxStride, int idxOff,
               _Float16* __restrict__ Ah, int M, int K, int Kpad)
{
  const long t = (long)blockIdx.x * blockDim.x + threadIdx.x;
  if (t >= (long)M * Kpad) return;
  const int m = (int)(t / Kpad), k = (int)(t % Kpad);
  const long row = rowIdx ? (long)rowIdx[(long)m * idxStride + idxOff] : (long)m;
  Ah[t] = (k < K) ? (_Float16)A[row * (long)lda + colOff + k] : (_Float16)0.f;
}

// Pre-pass 2: weights f32 KxN -> f16 transposed Npad x Kpad, zero-padded.
__global__ __launch_bounds__(256)
void cvt_w_f16(const float* __restrict__ W, int K, int N,
               _Float16* __restrict__ WT, int Kpad, int Npad)
{
  const int t = blockIdx.x * blockDim.x + threadIdx.x;
  if (t >= Npad * Kpad) return;
  const int n = t / Kpad, k = t % Kpad;
  WT[t] = (n < N && k < K) ? (_Float16)W[(long)k * N + n] : (_Float16)0.f;
}

// =====================================================================
// WMMA GEMM on pre-packed f16: Y = Ah(MxKpad) * WT^T(+bias). One wave
// computes a 64x16 strip (4 M-tiles). Per 64-wide K-step: issue ALL
// 20 b128 fragment loads first, then 8 v_wmma as 4 interleaved
// accumulation chains (lets scheduler use partial s_wait_loadcnt and
// overlap VMEM with the matrix pipe).
// =====================================================================
__global__ __launch_bounds__(256)
void gemm_wmma_pk(const _Float16* __restrict__ Ah,
                  const _Float16* __restrict__ WT,
                  const float* __restrict__ bias,
                  float* __restrict__ Y, int ldy,
                  int M, int N, int Kpad)
{
  const int wave    = blockIdx.x * (blockDim.x >> 5) + (threadIdx.x >> 5);
  const int tilesN  = (N + 15) >> 4;
  const int tilesM4 = (M + 63) >> 6;
  if (wave >= tilesM4 * tilesN) return;         // wave-uniform: EXEC stays full
  const int tm4  = wave / tilesN;
  const int tn   = wave % tilesN;
  const int lane = threadIdx.x & 31;
  const int half = lane >> 4;
  const int lr   = lane & 15;

  const _Float16* __restrict__ pa[4];
#pragma unroll
  for (int i = 0; i < 4; ++i) {
    int row = tm4 * 64 + i * 16 + lr;
    if (row > M - 1) row = M - 1;               // clamped load, store-guarded
    pa[i] = Ah + (long)row * Kpad;
  }
  const _Float16* __restrict__ pb = WT + (long)(tn * 16 + lr) * Kpad; // padded rows

  v8f acc[4] = {};
  for (int k0 = 0; k0 < Kpad; k0 += 64) {       // Kpad is a multiple of 64
    // ---- issue all fragment loads for this K-step first ----
    const v16h b0 = *(const v16h*)(pb + k0 + half * 16);
    const v16h b1 = *(const v16h*)(pb + k0 + 32 + half * 16);
    v16h a0[4], a1[4];
#pragma unroll
    for (int i = 0; i < 4; ++i) {
      const v8h x0 = *(const v8h*)(pa[i] + k0 + half * 16);
      const v8h x1 = *(const v8h*)(pa[i] + k0 + 8 + half * 16);
      a0[i] = __builtin_shufflevector(x0, x1, 0, 1, 2, 3, 4, 5, 6, 7,
                                      8, 9, 10, 11, 12, 13, 14, 15);
      const v8h y0 = *(const v8h*)(pa[i] + k0 + 32 + half * 16);
      const v8h y1 = *(const v8h*)(pa[i] + k0 + 40 + half * 16);
      a1[i] = __builtin_shufflevector(y0, y1, 0, 1, 2, 3, 4, 5, 6, 7,
                                      8, 9, 10, 11, 12, 13, 14, 15);
      __builtin_prefetch(pa[i] + k0 + 128, 0, 1);        // global_prefetch_b8
    }
    __builtin_prefetch(pb + k0 + 128, 0, 1);
    // ---- 8 WMMAs: 4 independent chains, dependent ops 4 apart ----
#pragma unroll
    for (int i = 0; i < 4; ++i)
      acc[i] = __builtin_amdgcn_wmma_f32_16x16x32_f16(false, a0[i], false, b0,
                                                      (short)0, acc[i], false, false);
#pragma unroll
    for (int i = 0; i < 4; ++i)
      acc[i] = __builtin_amdgcn_wmma_f32_16x16x32_f16(false, a1[i], false, b1,
                                                      (short)0, acc[i], false, false);
  }

  const int n = tn * 16 + lr;
  const float bz = (bias && n < N) ? bias[n] : 0.f;
#pragma unroll
  for (int i = 0; i < 4; ++i) {
#pragma unroll
    for (int r = 0; r < 8; ++r) {
      const int m = tm4 * 64 + i * 16 + r + 8 * half;     // C/D VGPR layout
      if (m < M && n < N)
        Y[(long)m * ldy + n] = acc[i][r] + bz;
    }
  }
}

// ----------------- column sum / sumsq reduction (for BN) -----------------
__global__ __launch_bounds__(256)
void col_stats(const float* __restrict__ Y, int M, int C, int ldy,
               float* __restrict__ stats /* [0..C)=sum, [C..2C)=sumsq */)
{
  __shared__ float ss[256], sq[256];
  const int c = blockIdx.x;
  float s = 0.f, q = 0.f;
  for (long m = (long)blockIdx.y * blockDim.x + threadIdx.x; m < M;
       m += (long)gridDim.y * blockDim.x) {
    float v = Y[m * ldy + c];
    s += v; q += v * v;
  }
  ss[threadIdx.x] = s; sq[threadIdx.x] = q;
  __syncthreads();
  for (int o = 128; o > 0; o >>= 1) {
    if (threadIdx.x < o) { ss[threadIdx.x] += ss[threadIdx.x + o];
                           sq[threadIdx.x] += sq[threadIdx.x + o]; }
    __syncthreads();
  }
  if (threadIdx.x == 0) {
    atomicAdd(&stats[c],     ss[0]);
    atomicAdd(&stats[C + c], sq[0]);
  }
}

// ----------------- BN + leaky-relu, optional accumulate, strided out -----------------
__global__ __launch_bounds__(256)
void bn_lrelu_k(const float* __restrict__ Y, int M, int C, int ldy,
                const float* __restrict__ stats,
                const float* __restrict__ g, const float* __restrict__ be,
                float slope, float invM,
                float* __restrict__ Out, int ldo, int colOff, int accum)
{
  const long i = (long)blockIdx.x * blockDim.x + threadIdx.x;
  if (i >= (long)M * C) return;
  const int m = (int)(i / C), c = (int)(i % C);
  const float mu  = stats[c] * invM;
  const float var = stats[C + c] * invM - mu * mu;
  float v = (Y[(long)m * ldy + c] - mu) * rsqrtf(var + BN_EPS) * g[c] + be[c];
  v = lrelu(v, slope);
  float* o = &Out[(long)m * ldo + colOff + c];
  if (accum) *o += v; else *o = v;
}

// ----------------- small utility kernels -----------------
__global__ void fill_k(float* p, long n, float v) {
  long i = (long)blockIdx.x * blockDim.x + threadIdx.x;
  if (i < n) p[i] = v;
}
__global__ void h1_init_k(const float* __restrict__ bias, float* __restrict__ h1) {
  long i = (long)blockIdx.x * blockDim.x + threadIdx.x;
  if (i < (long)NNODES * GAT_F) h1[i] = bias[i & (GAT_F - 1)];
}
__global__ void fix_max_k(float* m, int n) {
  int i = blockIdx.x * blockDim.x + threadIdx.x;
  if (i < n) { float v = m[i]; m[i] = (v * 0.f == 0.f) ? v : 0.f; } // -inf -> 0
}

// ----------------- GAT edge kernels -----------------
__global__ __launch_bounds__(256)
void gat_logits_k(const float* __restrict__ xl, const float* __restrict__ xr,
                  const int* __restrict__ ei, const float* __restrict__ att,
                  float* __restrict__ logits, float* __restrict__ segmax)
{
  const int t = blockIdx.x * blockDim.x + threadIdx.x;
  if (t >= E2 * NHEAD) return;
  const int e = t >> 2, h = t & 3;
  const int eb = (e < NEDGES) ? e : e - NEDGES;
  const int src = ei[eb * 2 + 0], dst = ei[eb * 2 + 1];
  const float* pl = xl + (long)src * GAT_F + h * HDIM;
  const float* pr = xr + (long)dst * GAT_F + h * HDIM;
  const float* pa = att + h * HDIM;
  float acc = 0.f;
#pragma unroll 8
  for (int d = 0; d < HDIM; ++d) acc += lrelu(pl[d] + pr[d], 0.2f) * pa[d];
  logits[t] = acc;
  atomicMaxF(&segmax[dst * NHEAD + h], acc);
}

__global__ __launch_bounds__(256)
void gat_exp_k(float* __restrict__ logits, const int* __restrict__ ei,
               const float* __restrict__ segmax, float* __restrict__ segsum)
{
  const int t = blockIdx.x * blockDim.x + threadIdx.x;
  if (t >= E2 * NHEAD) return;
  const int e = t >> 2, h = t & 3;
  const int eb = (e < NEDGES) ? e : e - NEDGES;
  const int dst = ei[eb * 2 + 1];
  const float ex = __expf(logits[t] - segmax[dst * NHEAD + h]);
  logits[t] = ex;                       // reuse buffer for ex
  atomicAdd(&segsum[dst * NHEAD + h], ex);
}

__global__ __launch_bounds__(256)
void gat_alpha_k(const float* __restrict__ ex, const int* __restrict__ ei,
                 const float* __restrict__ segsum, float* __restrict__ attnOut)
{
  const int t = blockIdx.x * blockDim.x + threadIdx.x;
  if (t >= E2 * NHEAD) return;
  const int e = t >> 2, h = t & 3;
  const int eb = (e < NEDGES) ? e : e - NEDGES;
  const int dst = ei[eb * 2 + 1];
  attnOut[t] = ex[t] / (segsum[dst * NHEAD + h] + 1e-16f);
}

__global__ __launch_bounds__(256)
void gat_scatter_k(const float* __restrict__ xl, const int* __restrict__ ei,
                   const float* __restrict__ alpha, float* __restrict__ h1)
{
  const long t = (long)blockIdx.x * blockDim.x + threadIdx.x;
  if (t >= (long)E2 * GAT_F) return;
  const long e = t >> 8;
  const int d = (int)(t & (GAT_F - 1));
  const int h = d >> 6;
  const int eb = (e < NEDGES) ? (int)e : (int)(e - NEDGES);
  const int src = ei[eb * 2 + 0], dst = ei[eb * 2 + 1];
  atomicAdd(&h1[(long)dst * GAT_F + d], alpha[e * NHEAD + h] * xl[(long)src * GAT_F + d]);
}

// ----------------- final 32->1 dot + sigmoid -----------------
__global__ __launch_bounds__(256)
void fc3_sigmoid_k(const float* __restrict__ F2, const float* __restrict__ W3,
                   const float* __restrict__ b3, float* __restrict__ out)
{
  const int e = blockIdx.x * blockDim.x + threadIdx.x;
  if (e >= NEDGES) return;
  float acc = b3[0];
#pragma unroll
  for (int k = 0; k < 32; ++k) acc += F2[(long)e * 32 + k] * W3[k];
  out[e] = 1.f / (1.f + __expf(-acc));
}

// =====================================================================
// Host-side orchestration
// =====================================================================
static inline void launch_gemm(hipStream_t s, const float* A, int lda, int colOff,
                               const int* idx, int istride, int ioff,
                               const float* W, const float* b,
                               float* Y, int ldy, int M, int N, int K,
                               _Float16* Ah, _Float16* Wh)
{
  const int Kpad = (K + 63) & ~63;
  const int Npad = (N + 15) & ~15;
  // pack weights (tiny) and activations (streaming) to f16
  cvt_w_f16<<<(Npad * Kpad + 255) / 256, 256, 0, s>>>(W, K, N, Wh, Kpad, Npad);
  const long na = (long)M * Kpad;
  cvt_a_f16<<<(int)((na + 255) / 256), 256, 0, s>>>(A, lda, colOff, idx, istride, ioff,
                                                    Ah, M, K, Kpad);
  const int tiles  = ((M + 63) / 64) * (Npad / 16);
  const int blocks = (tiles + 7) / 8;           // 8 waves (256 threads) per block
  gemm_wmma_pk<<<blocks, 256, 0, s>>>(Ah, Wh, b, Y, ldy, M, N, Kpad);
}

static inline void bn_stage(hipStream_t s, float* stats,
                            const float* Y, int M, int C, int ldy,
                            const float* g, const float* be, float slope,
                            float* Out, int ldo, int colOff, int accum)
{
  hipMemsetAsync(stats, 0, 2 * C * sizeof(float), s);
  dim3 gs(C, 128);
  col_stats<<<gs, 256, 0, s>>>(Y, M, C, ldy, stats);
  const long tot = (long)M * C;
  bn_lrelu_k<<<(int)((tot + 255) / 256), 256, 0, s>>>(Y, M, C, ldy, stats, g, be,
                                                      slope, 1.f / (float)M,
                                                      Out, ldo, colOff, accum);
}

struct SubP { const float *W1, *W2, *b1, *b2, *be1, *be2, *g1, *g2; };
static inline SubP subp(void* const* d, int base) {
  SubP p;
  p.W1  = (const float*)d[base + 0]; p.W2  = (const float*)d[base + 1];
  p.b1  = (const float*)d[base + 2]; p.b2  = (const float*)d[base + 3];
  p.be1 = (const float*)d[base + 4]; p.be2 = (const float*)d[base + 5];
  p.g1  = (const float*)d[base + 6]; p.g2  = (const float*)d[base + 7];
  return p;
}

// _sub_fwd: X(:,colOff:colOff+K) @ W1 -> bn/lrelu -> @ W2 -> bn/lrelu -> Out[:,outOff:+64]
static inline void run_sub(hipStream_t s, const SubP& p,
                           const float* X, int ldx, int colOff, int M, int K,
                           float* Yt /*>=M*64*/, float* Hmid /*M*32*/,
                           float* Out, int ldo, int outOff, float* stats,
                           _Float16* Ah, _Float16* Wh)
{
  launch_gemm(s, X, ldx, colOff, nullptr, 0, 0, p.W1, p.b1, Yt, 32, M, 32, K, Ah, Wh);
  bn_stage(s, stats, Yt, M, 32, 32, p.g1, p.be1, 0.01f, Hmid, 32, 0, 0);
  launch_gemm(s, Hmid, 32, 0, nullptr, 0, 0, p.W2, p.b2, Yt, 64, M, 64, 32, Ah, Wh);
  bn_stage(s, stats, Yt, M, 64, 64, p.g2, p.be2, 0.01f, Out, ldo, outOff, 0);
}

extern "C" void kernel_launch(void* const* d_in, const int* in_sizes, int n_in,
                              void* d_out, int out_size, void* d_ws, size_t ws_size,
                              hipStream_t stream)
{
  (void)in_sizes; (void)n_in; (void)out_size; (void)ws_size;
  const float* interaction = (const float*)d_in[0];  // E x 16
  const float* graph       = (const float*)d_in[1];  // N x 448
  const int*   ei          = (const int*)  d_in[2];  // E x 2

  // params flattened leaf-wise, dict keys sorted (jax pytree order):
  const SubP co_loc = subp(d_in, 3);                 // 3..10
  const SubP co_p   = subp(d_in, 11);                // 11..18
  const float* fW1 = (const float*)d_in[19], *fW2 = (const float*)d_in[20],
             * fW3 = (const float*)d_in[21], *fb1 = (const float*)d_in[22],
             * fb2 = (const float*)d_in[23], *fb3 = (const float*)d_in[24],
             * fbe1= (const float*)d_in[25], *fbe2= (const float*)d_in[26],
             * fg1 = (const float*)d_in[27], *fg2 = (const float*)d_in[28];
  const SubP g_head = subp(d_in, 29);                // 29..36
  const float* Wl  = (const float*)d_in[37], *Wr  = (const float*)d_in[38],
             * att = (const float*)d_in[39], *gbias=(const float*)d_in[40],
             * bl  = (const float*)d_in[41], *br  = (const float*)d_in[42];
  const SubP l_head = subp(d_in, 43);                // 43..50
  const SubP m_head = subp(d_in, 51);                // 51..58
  const SubP p_head = subp(d_in, 59);                // 59..66
  const float* phiW = (const float*)d_in[67], *phib = (const float*)d_in[68],
             * phibe= (const float*)d_in[69], *phig = (const float*)d_in[70];
  const float* rW1 = (const float*)d_in[71], *rW2 = (const float*)d_in[72],
             * rb1 = (const float*)d_in[73], *rb2 = (const float*)d_in[74],
             * rbe1= (const float*)d_in[75], *rbe2= (const float*)d_in[76],
             * rg1 = (const float*)d_in[77], *rg2 = (const float*)d_in[78];

  // -------- workspace layout (floats; all region sizes are multiples of 64) --------
  float* w = (float*)d_ws;
  float* xl     = w; w += (long)NNODES * GAT_F;      // 12.8M
  float* xr     = w; w += (long)NNODES * GAT_F;      // 12.8M
  float* h1     = w; w += (long)NNODES * GAT_F;      // 12.8M
  float* gm     = w; w += (long)NNODES * 192;        // 9.6M
  float* Hn32   = w; w += (long)NNODES * 32;         // 1.6M
  float* Ytn    = w; w += (long)NNODES * 64;         // 3.2M
  float* logits = w; w += (long)E2 * NHEAD;          // 1.6M
  float* segmax = w; w += NNODES * NHEAD;
  float* segsum = w; w += NNODES * NHEAD;
  float* stats  = w; w += 512;
  float* preds1 = w; w += (long)NEDGES * 128;        // 25.6M
  float* R1     = w; w += (long)NEDGES * 128;        // 25.6M
  float* z      = w; w += (long)NEDGES * 256;        // 51.2M
  float* C32e   = w; w += (long)NEDGES * 32;
  float* F1     = w; w += (long)NEDGES * 64;
  float* F2     = w; w += (long)NEDGES * 32;
  float* Ye32   = w; w += (long)NEDGES * 32;
  _Float16* Ah  = (_Float16*)w; w += (long)NEDGES * 128;  // E*256 halfs (f16 staging)
  _Float16* Wh  = (_Float16*)w; w += (256 * 256) / 2;     // 256x256 halfs
  // aliases: GAT-phase buffers are dead by the time these are used
  float* Ye128  = xl;   // E*128 fits exactly in xl+xr (first use after gat_scatter)
  float* Ye64   = gm;   // E*64 fits in gm+Hn32+Ytn (first use after xl/xr gemms)

  float* predOut = (float*)d_out;            // E sigmoid outputs
  float* attnOut = (float*)d_out + NEDGES;   // E2 x NHEAD alphas

  // -------- node feature heads -> gm (N x 192) --------
  run_sub(stream, g_head, graph, 448, 0,   NNODES, 256, Ytn, Hn32, gm, 192, 0,   stats, Ah, Wh);
  run_sub(stream, p_head, graph, 448, 256, NNODES, 128, Ytn, Hn32, gm, 192, 64,  stats, Ah, Wh);
  run_sub(stream, l_head, graph, 448, 384, NNODES, 64,  Ytn, Hn32, gm, 192, 128, stats, Ah, Wh);

  // -------- GAT linear: xl/xr = gm @ Wl/Wr + b  (K=192 already mult. of 64) --------
  launch_gemm(stream, gm, 192, 0, nullptr, 0, 0, Wl, bl, xl, GAT_F, NNODES, GAT_F, 192, Ah, Wh);
  launch_gemm(stream, gm, 192, 0, nullptr, 0, 0, Wr, br, xr, GAT_F, NNODES, GAT_F, 192, Ah, Wh);

  // -------- GAT attention (segmented softmax over dst) --------
  {
    const int nsm = NNODES * NHEAD;
    fill_k<<<(nsm + 255) / 256, 256, 0, stream>>>(segmax, nsm, -INFINITY);
    hipMemsetAsync(segsum, 0, nsm * sizeof(float), stream);
    const int nt = E2 * NHEAD;
    gat_logits_k<<<(nt + 255) / 256, 256, 0, stream>>>(xl, xr, ei, att, logits, segmax);
    fix_max_k<<<(nsm + 255) / 256, 256, 0, stream>>>(segmax, nsm);
    gat_exp_k<<<(nt + 255) / 256, 256, 0, stream>>>(logits, ei, segmax, segsum);
    gat_alpha_k<<<(nt + 255) / 256, 256, 0, stream>>>(logits, ei, segsum, attnOut);
    const long nh1 = (long)NNODES * GAT_F;
    h1_init_k<<<(int)((nh1 + 255) / 256), 256, 0, stream>>>(gbias, h1);
    const long nsc = (long)E2 * GAT_F;
    gat_scatter_k<<<(int)((nsc + 255) / 256), 256, 0, stream>>>(xl, ei, attnOut, h1);
  }

  // -------- phi on both edge endpoints (gather fused in cvt_a), preds1 = phi(a)+phi(b) --------
  launch_gemm(stream, h1, GAT_F, 0, ei, 2, 0, phiW, phib, Ye128, 128, NEDGES, 128, GAT_F, Ah, Wh);
  bn_stage(stream, stats, Ye128, NEDGES, 128, 128, phig, phibe, 0.01f, preds1, 128, 0, 0);
  launch_gemm(stream, h1, GAT_F, 0, ei, 2, 1, phiW, phib, Ye128, 128, NEDGES, 128, GAT_F, Ah, Wh);
  bn_stage(stream, stats, Ye128, NEDGES, 128, 128, phig, phibe, 0.01f, preds1, 128, 0, 1);

  // -------- rho1: 128->128->64, result into z[:,0:64] --------
  launch_gemm(stream, preds1, 128, 0, nullptr, 0, 0, rW1, rb1, Ye128, 128, NEDGES, 128, 128, Ah, Wh);
  bn_stage(stream, stats, Ye128, NEDGES, 128, 128, rg1, rbe1, 0.01f, R1, 128, 0, 0);
  launch_gemm(stream, R1, 128, 0, nullptr, 0, 0, rW2, rb2, Ye64, 64, NEDGES, 64, 128, Ah, Wh);
  bn_stage(stream, stats, Ye64, NEDGES, 64, 64, rg2, rbe2, 0.01f, z, 256, 0, 0);

  // -------- interaction heads -> z[:,64:256] --------
  run_sub(stream, co_p,   interaction, 16, 0, NEDGES, 1,  Ye64, C32e, z, 256, 64,  stats, Ah, Wh);
  run_sub(stream, co_loc, interaction, 16, 1, NEDGES, 1,  Ye64, C32e, z, 256, 128, stats, Ah, Wh);
  run_sub(stream, m_head, interaction, 16, 2, NEDGES, 14, Ye64, C32e, z, 256, 192, stats, Ah, Wh);

  // -------- fc: 256->64->32->1 + sigmoid --------
  launch_gemm(stream, z, 256, 0, nullptr, 0, 0, fW1, fb1, Ye64, 64, NEDGES, 64, 256, Ah, Wh);
  bn_stage(stream, stats, Ye64, NEDGES, 64, 64, fg1, fbe1, 0.01f, F1, 64, 0, 0);
  launch_gemm(stream, F1, 64, 0, nullptr, 0, 0, fW2, fb2, Ye32, 32, NEDGES, 32, 64, Ah, Wh);
  bn_stage(stream, stats, Ye32, NEDGES, 32, 32, fg2, fbe2, 0.01f, F2, 32, 0, 0);
  fc3_sigmoid_k<<<(NEDGES + 255) / 256, 256, 0, stream>>>(F2, fW3, fb3, predOut);
}
